// Attention_44504451121208
// MI455X (gfx1250) — compile-verified, matrix operand order checked by reading, not run on hardware
//
#include <hip/hip_runtime.h>
#include <hip/hip_bf16.h>
#include <stdint.h>

// ---------------------------------------------------------------------------
// Types for CDNA5 WMMA (wave32): 16x16x32 bf16 -> f32
// ---------------------------------------------------------------------------
typedef __attribute__((ext_vector_type(16))) __bf16 v16bf;
typedef __attribute__((ext_vector_type(8)))  __bf16 v8bf;
typedef __attribute__((ext_vector_type(8)))  float  v8f;

union Frag16 { v16bf v; struct { v8bf lo, hi; } s; };

__device__ __forceinline__ v8f wmma_bf16(v16bf a, v16bf b, v8f c) {
  return __builtin_amdgcn_wmma_f32_16x16x32_bf16(
      /*neg_a=*/false, a, /*neg_b=*/false, b,
      /*c_mod=*/(short)0, c, /*reuse_a=*/false, /*reuse_b=*/false);
}

// Load a 16x32 bf16 WMMA operand fragment (A layout; B mirrored with
// n = lane&15) from a row-major tile with row stride `stride` (elements).
// ISA 7.12.2: lane L holds row L&15; elems 0..7 = k kb..kb+7,
// elems 8..15 = k 16+kb.., kb = (L>>4)*8.  Two ds_load_b128 per fragment.
__device__ __forceinline__ v16bf load_frag(const __bf16* tile, int stride, int lane) {
  const int r  = lane & 15;
  const int kb = (lane >> 4) << 3;
  const __bf16* p = tile + r * stride + kb;
  Frag16 u;
  u.s.lo = *(const v8bf*)(p);
  u.s.hi = *(const v8bf*)(p + 16);
  return u.v;
}

// ---------------------------------------------------------------------------
// CDNA5 async global->LDS copy (ASYNCcnt-tracked). The builtin expects
// 'int __vector(4) addrspace(1)*' / LDS pointer operands; build them through
// uintptr_t (generic->global is identity; generic LDS pointers carry the LDS
// offset in their low 32 bits per the flat-addressing rule).
// ---------------------------------------------------------------------------
typedef int v4i_vs __attribute__((vector_size(16)));
typedef __attribute__((address_space(1))) v4i_vs* gptr_v4i;  // global pointee
typedef __attribute__((address_space(3))) v4i_vs* lptr_v4i;  // LDS pointee

#if defined(__has_builtin) && __has_builtin(__builtin_amdgcn_global_load_async_to_lds_b128)
#define HAS_ASYNC_LDS 1
__device__ __forceinline__ void async_cp16(const void* g, void* l) {
  __builtin_amdgcn_global_load_async_to_lds_b128(
      (gptr_v4i)(uintptr_t)g,
      (lptr_v4i)(unsigned int)(uintptr_t)l, 0, 0);
}
#else
#define HAS_ASYNC_LDS 0
__device__ __forceinline__ void async_cp16(const void* g, void* l) {
  *(v8bf*)l = *(const v8bf*)g;
}
#endif

#if defined(__has_builtin) && __has_builtin(__builtin_amdgcn_s_wait_asynccnt)
#define WAIT_ASYNC0() __builtin_amdgcn_s_wait_asynccnt(0)
#elif HAS_ASYNC_LDS
#define WAIT_ASYNC0() asm volatile("s_wait_asynccnt 0x0" ::: "memory")
#else
#define WAIT_ASYNC0() do {} while (0)
#endif

// ---------------------------------------------------------------------------
// GEMM: C[M,N] = A[M,K] @ W[N,K]^T (+bias). A is f32 or bf16; W always f32
// (original weights) converted to bf16 on the way into LDS.
// Block tile 128x256, BK=32, 256 threads = 8 waves, wave tile 64x64:
// 16 WMMAs per 8 fragment loads per k-step.
// ---------------------------------------------------------------------------
#define GBM 128
#define GBN 256
#define GBK 32
#define GLD (GBK + 8)   // padded LDS row stride (bf16); 80B rows, 16B aligned

template<bool A_F32, bool OUT_F32>
__global__ __launch_bounds__(256)
void gemm_wmma(const void* __restrict__ Ap, const float* __restrict__ W,
               const float* __restrict__ bias, void* __restrict__ Cp,
               int M, int N, int K)
{
  __shared__ __attribute__((aligned(16))) __bf16 As[GBM][GLD];   // 10 KB
  __shared__ __attribute__((aligned(16))) __bf16 Bs[GBN][GLD];   // 20 KB

  const int tid  = threadIdx.x;
  const int lane = tid & 31;
  const int wid  = tid >> 5;
  const int wm   = wid & 1;    // 2 wave rows  (64 each)
  const int wn   = wid >> 1;   // 4 wave cols  (64 each)
  const int bm0  = blockIdx.y * GBM;
  const int bn0  = blockIdx.x * GBN;

  const float*  Af = (const float*)Ap;
  const __bf16* Ab = (const __bf16*)Ap;

  v8f acc[4][4];
#pragma unroll
  for (int mi = 0; mi < 4; ++mi)
#pragma unroll
    for (int ni = 0; ni < 4; ++ni)
#pragma unroll
      for (int v = 0; v < 8; ++v) acc[mi][ni][v] = 0.0f;

  // stage-in mapping: thread t owns 16 contiguous k of row t>>1
  const int lrow = tid >> 1;
  const int lcol = (tid & 1) * 16;

  for (int k0 = 0; k0 < K; k0 += GBK) {
    __syncthreads();
    // ---- A tile -> LDS ----
    if (A_F32) {
      const float* src = Af + (size_t)(bm0 + lrow) * K + k0 + lcol;
      float4 f0 = *(const float4*)(src + 0);
      float4 f1 = *(const float4*)(src + 4);
      float4 f2 = *(const float4*)(src + 8);
      float4 f3 = *(const float4*)(src + 12);
      __bf16* d = &As[lrow][lcol];
      d[0]=(__bf16)f0.x; d[1]=(__bf16)f0.y; d[2]=(__bf16)f0.z; d[3]=(__bf16)f0.w;
      d[4]=(__bf16)f1.x; d[5]=(__bf16)f1.y; d[6]=(__bf16)f1.z; d[7]=(__bf16)f1.w;
      d[8]=(__bf16)f2.x; d[9]=(__bf16)f2.y; d[10]=(__bf16)f2.z; d[11]=(__bf16)f2.w;
      d[12]=(__bf16)f3.x; d[13]=(__bf16)f3.y; d[14]=(__bf16)f3.z; d[15]=(__bf16)f3.w;
    } else {
      // bf16 -> bf16 straight move: async DMA to LDS on CDNA5
      const __bf16* src = Ab + (size_t)(bm0 + lrow) * K + k0 + lcol;
      async_cp16(src,     &As[lrow][lcol]);
      async_cp16(src + 8, &As[lrow][lcol + 8]);
    }
    // ---- W tile -> LDS (2 row-segments of 128; convert f32->bf16) ----
#pragma unroll
    for (int seg = 0; seg < 2; ++seg) {
      const int brow = lrow + seg * 128;
      const float* src = W + (size_t)(bn0 + brow) * K + k0 + lcol;
      float4 f0 = *(const float4*)(src + 0);
      float4 f1 = *(const float4*)(src + 4);
      float4 f2 = *(const float4*)(src + 8);
      float4 f3 = *(const float4*)(src + 12);
      __bf16* d = &Bs[brow][lcol];
      d[0]=(__bf16)f0.x; d[1]=(__bf16)f0.y; d[2]=(__bf16)f0.z; d[3]=(__bf16)f0.w;
      d[4]=(__bf16)f1.x; d[5]=(__bf16)f1.y; d[6]=(__bf16)f1.z; d[7]=(__bf16)f1.w;
      d[8]=(__bf16)f2.x; d[9]=(__bf16)f2.y; d[10]=(__bf16)f2.z; d[11]=(__bf16)f2.w;
      d[12]=(__bf16)f3.x; d[13]=(__bf16)f3.y; d[14]=(__bf16)f3.z; d[15]=(__bf16)f3.w;
    }
    // prefetch next A k-tile into cache
    if (k0 + GBK < K) {
      if (A_F32) __builtin_prefetch(Af + (size_t)(bm0 + lrow) * K + k0 + GBK + lcol, 0, 3);
      else       __builtin_prefetch(Ab + (size_t)(bm0 + lrow) * K + k0 + GBK + lcol, 0, 3);
    }
    if (!A_F32) WAIT_ASYNC0();
    __syncthreads();

    // ---- 16x16x32 WMMA micro-kernel: 64x64 per wave ----
    v16bf a[4], b[4];
#pragma unroll
    for (int mi = 0; mi < 4; ++mi)
      a[mi] = load_frag(&As[wm * 64 + mi * 16][0], GLD, lane);
#pragma unroll
    for (int ni = 0; ni < 4; ++ni)
      b[ni] = load_frag(&Bs[wn * 64 + ni * 16][0], GLD, lane);
#pragma unroll
    for (int mi = 0; mi < 4; ++mi)
#pragma unroll
      for (int ni = 0; ni < 4; ++ni)
        acc[mi][ni] = wmma_bf16(a[mi], b[ni], acc[mi][ni]);
  }

  // ---- epilogue: C-fragment row = v + 8*(lane>>4), col = lane&15 ----
  const int half  = lane >> 4;
  const int col16 = lane & 15;
#pragma unroll
  for (int ni = 0; ni < 4; ++ni) {
    const int col = bn0 + wn * 64 + ni * 16 + col16;
    const float bv = bias ? bias[col] : 0.0f;
#pragma unroll
    for (int mi = 0; mi < 4; ++mi)
#pragma unroll
      for (int v = 0; v < 8; ++v) {
        const int row = bm0 + wm * 64 + mi * 16 + half * 8 + v;
        const float val = acc[mi][ni][v] + bv;
        if (OUT_F32) ((float*)Cp)[(size_t)row * N + col] = val;
        else         ((__bf16*)Cp)[(size_t)row * N + col] = (__bf16)val;
      }
  }
}

// ---------------------------------------------------------------------------
// Flash attention over packed QKV [B*384][2304] (bf16), head dim 64.
// Row layout: q at h*64, k at 768+h*64, v at 1536+h*64.
// 128 threads = 4 waves; each wave owns 16 q rows; kv tiled by 64.
// grid = (q_tiles_of_64, H, B). Writes bf16 rows into the concat buffer(s).
// ---------------------------------------------------------------------------
#define AKV 64
#define ADH 64
#define AST (ADH + 8)   // 72-elem row stride: 144B (16B-multiple)

__global__ __launch_bounds__(128)
void attn_wmma(const __bf16* __restrict__ QKV,
               int q_row0, int kv_row0, int kv_len,
               __bf16* __restrict__ Y1, __bf16* __restrict__ Y2,
               int out_row0)
{
  __shared__ __attribute__((aligned(16))) __bf16 Ks[AKV][AST];    // [kv][d]
  __shared__ __attribute__((aligned(16))) __bf16 Vt[ADH][AST];    // [d][kv]
  __shared__ __attribute__((aligned(16))) __bf16 Ps[4][16][AST];  // per-wave P

  const int tid  = threadIdx.x;
  const int lane = tid & 31;
  const int wid  = tid >> 5;
  const int qt = blockIdx.x;
  const int h  = blockIdx.y;
  const int b  = blockIdx.z;
  const int browbase = b * 384;
  const float scale = 0.125f;   // Dh^-0.5, Dh = 64

  // ---- Q fragments straight from global (A layout) ----
  v16bf aq[2];
  {
    const int qrow = browbase + q_row0 + qt * 64 + wid * 16;
    const __bf16* qb = QKV + (size_t)qrow * 2304 + h * 64;
    const int r  = lane & 15;
    const int kb = (lane >> 4) << 3;
    const __bf16* p = qb + (size_t)r * 2304 + kb;
    Frag16 u0, u1;
    u0.s.lo = *(const v8bf*)(p);        u0.s.hi = *(const v8bf*)(p + 16);
    u1.s.lo = *(const v8bf*)(p + 32);   u1.s.hi = *(const v8bf*)(p + 48);
    aq[0] = u0.v; aq[1] = u1.v;
  }

  v8f o[4];
  float mrow[8], lsum[8];
#pragma unroll
  for (int j = 0; j < 4; ++j)
#pragma unroll
    for (int v = 0; v < 8; ++v) o[j][v] = 0.0f;
#pragma unroll
  for (int v = 0; v < 8; ++v) { mrow[v] = -1e30f; lsum[v] = 0.0f; }

  const int nkv = kv_len / AKV;
  for (int t = 0; t < nkv; ++t) {
    __syncthreads();
    // ---- stage K (async DMA) and V (manual transpose) tiles to LDS ----
    {
      const int r   = tid >> 1;          // kv row 0..63
      const int seg = (tid & 1) * 32;    // d segment
      const size_t grow = (size_t)(browbase + kv_row0 + t * AKV + r) * 2304 + h * 64 + seg;
      const __bf16* kp = QKV + grow + 768;
      async_cp16(kp,      &Ks[r][seg]);
      async_cp16(kp + 8,  &Ks[r][seg + 8]);
      async_cp16(kp + 16, &Ks[r][seg + 16]);
      async_cp16(kp + 24, &Ks[r][seg + 24]);
      const __bf16* vp = QKV + grow + 1536;
      v8bf v0 = *(const v8bf*)(vp);
      v8bf v1 = *(const v8bf*)(vp + 8);
      v8bf v2 = *(const v8bf*)(vp + 16);
      v8bf v3 = *(const v8bf*)(vp + 24);
#pragma unroll
      for (int i = 0; i < 8; ++i) Vt[seg + i][r]      = v0[i];
#pragma unroll
      for (int i = 0; i < 8; ++i) Vt[seg + 8 + i][r]  = v1[i];
#pragma unroll
      for (int i = 0; i < 8; ++i) Vt[seg + 16 + i][r] = v2[i];
#pragma unroll
      for (int i = 0; i < 8; ++i) Vt[seg + 24 + i][r] = v3[i];
    }
    WAIT_ASYNC0();
    __syncthreads();

    // ---- S = Q K^T : four 16x16 fragments over 64 kv columns ----
    v8f s[4];
#pragma unroll
    for (int ni = 0; ni < 4; ++ni) {
#pragma unroll
      for (int v = 0; v < 8; ++v) s[ni][v] = 0.0f;
      v16bf bk0 = load_frag(&Ks[ni * 16][0],  AST, lane);   // d 0..31
      v16bf bk1 = load_frag(&Ks[ni * 16][32], AST, lane);   // d 32..63
      s[ni] = wmma_bf16(aq[0], bk0, s[ni]);
      s[ni] = wmma_bf16(aq[1], bk1, s[ni]);
    }

    // ---- online softmax: rows map to (v, lane>>4); cols across 16 lanes ----
    float p[4][8], fac[8];
#pragma unroll
    for (int v = 0; v < 8; ++v) {
      float s0 = s[0][v] * scale, s1 = s[1][v] * scale;
      float s2 = s[2][v] * scale, s3 = s[3][v] * scale;
      float mx = fmaxf(fmaxf(s0, s1), fmaxf(s2, s3));
#pragma unroll
      for (int msk = 1; msk < 16; msk <<= 1) mx = fmaxf(mx, __shfl_xor(mx, msk, 32));
      float mnew = fmaxf(mrow[v], mx);
      float f = __expf(mrow[v] - mnew);
      p[0][v] = __expf(s0 - mnew);
      p[1][v] = __expf(s1 - mnew);
      p[2][v] = __expf(s2 - mnew);
      p[3][v] = __expf(s3 - mnew);
      float rs = (p[0][v] + p[1][v]) + (p[2][v] + p[3][v]);
#pragma unroll
      for (int msk = 1; msk < 16; msk <<= 1) rs += __shfl_xor(rs, msk, 32);
      lsum[v] = lsum[v] * f + rs;
      mrow[v] = mnew;
      fac[v]  = f;
    }
#pragma unroll
    for (int j = 0; j < 4; ++j)
#pragma unroll
      for (int v = 0; v < 8; ++v) o[j][v] *= fac[v];

    // ---- P -> bf16 via per-wave LDS staging, reload in A layout ----
    {
      const int half = lane >> 4, c = lane & 15;
#pragma unroll
      for (int v = 0; v < 8; ++v)
#pragma unroll
        for (int ni = 0; ni < 4; ++ni)
          Ps[wid][half * 8 + v][c + 16 * ni] = (__bf16)p[ni][v];
    }
    v16bf ap0 = load_frag(&Ps[wid][0][0],  AST, lane);   // kv 0..31
    v16bf ap1 = load_frag(&Ps[wid][0][32], AST, lane);   // kv 32..63

    // ---- O += P V (two k-steps of 32 kv) ----
#pragma unroll
    for (int j = 0; j < 4; ++j) {
      v16bf bv0 = load_frag(&Vt[j * 16][0],  AST, lane);
      v16bf bv1 = load_frag(&Vt[j * 16][32], AST, lane);
      o[j] = wmma_bf16(ap0, bv0, o[j]);
      o[j] = wmma_bf16(ap1, bv1, o[j]);
    }
  }

  // ---- normalize and store into concat buffer(s) ----
  {
    const int half = lane >> 4, c = lane & 15;
    const int orow = browbase + out_row0 + qt * 64 + wid * 16;
#pragma unroll
    for (int j = 0; j < 4; ++j) {
      const int col = h * 64 + j * 16 + c;
#pragma unroll
      for (int v = 0; v < 8; ++v) {
        const float val = o[j][v] / lsum[v];
        const size_t idx = (size_t)(orow + half * 8 + v) * 768 + col;
        Y1[idx] = (__bf16)val;
        if (Y2) Y2[idx] = (__bf16)val;
      }
    }
  }
}

// ---------------------------------------------------------------------------
// Launch: 2 QKV gemms -> 3 attention passes -> 2 projection gemms.
// ---------------------------------------------------------------------------
extern "C" void kernel_launch(void* const* d_in, const int* in_sizes, int n_in,
                              void* d_out, int out_size, void* d_ws, size_t ws_size,
                              hipStream_t stream) {
  (void)in_sizes; (void)n_in; (void)out_size; (void)ws_size;
  const float* x      = (const float*)d_in[0];
  const float* x_hsi  = (const float*)d_in[1];
  const float* qkv_w  = (const float*)d_in[2];   // [2304,768]
  const float* proj_w = (const float*)d_in[3];   // [768,768]
  const float* proj_b = (const float*)d_in[4];   // [768]

  const int B = 64, N = 384, C = 768, H = 12;
  const int M  = B * N;     // 24576
  const int C3 = 3 * C;     // 2304

  char* ws = (char*)d_ws;
  size_t off = 0;
  __bf16* qkvx = (__bf16*)(ws + off); off += (size_t)M * C3 * sizeof(__bf16);
  __bf16* qkvh = (__bf16*)(ws + off); off += (size_t)M * C3 * sizeof(__bf16);
  __bf16* Y    = (__bf16*)(ws + off); off += (size_t)M * C  * sizeof(__bf16);
  __bf16* Yh   = (__bf16*)(ws + off); off += (size_t)M * C  * sizeof(__bf16);

  dim3 gblk(256);
  dim3 g1(C3 / GBN, M / GBM);
  gemm_wmma<true, false><<<g1, gblk, 0, stream>>>(x,     qkv_w, nullptr, qkvx, M, C3, C);
  gemm_wmma<true, false><<<g1, gblk, 0, stream>>>(x_hsi, qkv_w, nullptr, qkvh, M, C3, C);

  dim3 ablk(128);
  // x_mt: q rows [0,128), kv rows [0,128); written to BOTH concat buffers.
  attn_wmma<<<dim3(2, H, B), ablk, 0, stream>>>(qkvx, 0,   0, 128, Y,  Yh,      0);
  // x_s: q rows [128,384), kv all 384.
  attn_wmma<<<dim3(4, H, B), ablk, 0, stream>>>(qkvx, 128, 0, 384, Y,  nullptr, 128);
  // x_s_hsi: q rows [128,384) of hsi, kv all 384 of hsi.
  attn_wmma<<<dim3(4, H, B), ablk, 0, stream>>>(qkvh, 128, 0, 384, Yh, nullptr, 128);

  float* out1 = (float*)d_out;
  float* out2 = out1 + (size_t)M * C;
  dim3 g2(C / GBN, M / GBM);
  gemm_wmma<false, true><<<g2, gblk, 0, stream>>>(Y,  proj_w, proj_b, out1, M, C, C);
  gemm_wmma<false, true><<<g2, gblk, 0, stream>>>(Yh, proj_w, proj_b, out2, M, C, C);
}
